// Policy_5480378269890
// MI455X (gfx1250) — compile-verified, matrix operand order checked by reading, not run on hardware
//
#include <hip/hip_runtime.h>

typedef __attribute__((ext_vector_type(16))) _Float16 v16h;
typedef __attribute__((ext_vector_type(8)))  float    v8f;

#define BN_EPS 1e-5f

// K index packing for 16-bit A/B fragments of V_WMMA_*_16X16X32_*:
// lane group (lane>=16) owns K={8..15,24..31}; halves j<8 -> low 16 K's, j>=8 -> high 16 K's.
__device__ __forceinline__ int kmap(int grp, int j) {
    return (j < 8) ? (grp * 8 + j) : (16 + grp * 8 + (j - 8));
}

__global__ __launch_bounds__(256) void naf_policy_kernel(
    const float* __restrict__ in0,                            // [2,32]
    const float* __restrict__ u,                              // [2,8]
    const float* __restrict__ w1c, const float* __restrict__ b1c,
    const float* __restrict__ w1o, const float* __restrict__ b1o,
    const float* __restrict__ w2c, const float* __restrict__ b2c,
    const float* __restrict__ w2o, const float* __restrict__ b2o,
    const float* __restrict__ w3c, const float* __restrict__ b3c,
    const float* __restrict__ w3o, const float* __restrict__ b3o,
    const float* __restrict__ g0,  const float* __restrict__ be0,
    const float* __restrict__ W1,  const float* __restrict__ bl1, // [128,32],[128]
    const float* __restrict__ W2,  const float* __restrict__ bl2, // [128,128],[128]
    const float* __restrict__ Vw,  const float* __restrict__ Vb,  // [1,128],[1]
    const float* __restrict__ Mw,  const float* __restrict__ Mb,  // [8,128],[8]
    const float* __restrict__ Lw,  const float* __restrict__ Lb,  // [64,128],[64]
    float* __restrict__ out)                                      // mu[16] | Q[2] | V[2]
{
    __shared__ float    s_conv[3][16][7];
    __shared__ float    s_bout[3][16];
    __shared__ float    s_x[2][32];
    __shared__ _Float16 s_actA[16][128];   // A tile (f16); rows 2..15 stay zero forever
    __shared__ _Float16 s_actB[16][128];   // ditto
    __shared__ float    s_h3[2][80];       // head pre-activations (bias included)

    const int tid  = threadIdx.x;
    const int lane = tid & 31;
    const int wv   = tid >> 5;   // wave id 0..7
    const int mrow = lane & 15;  // M row (A) / N col (B,C,D)
    const int grp  = lane >> 4;  // K half-group / M-high select

    // ---- warm L2 with the weight matrices (global_prefetch_b8) ----
    for (int i = tid; i < 512; i += 256) __builtin_prefetch(W2 + i * 32, 0, 3); // 64KB
    for (int i = tid; i < 256; i += 256) __builtin_prefetch(Lw + i * 32, 0, 3); // 32KB
    if (tid < 128) __builtin_prefetch(W1 + tid * 32, 0, 3);                     // 16KB
    if (tid < 32)  __builtin_prefetch(Mw + tid * 32, 0, 3);

    // zero-fill BOTH padded A tiles once; per-layer stores touch only rows 0,1
    for (int i = tid; i < 16 * 128; i += 256) {
        (&s_actA[0][0])[i] = (_Float16)0.0f;
        (&s_actB[0][0])[i] = (_Float16)0.0f;
    }
    __syncthreads();

    // ---- phase 0a: conv1d(2->16,k4,p1) + relu for 3 branches -> s_conv ----
    for (int idx = tid; idx < 3 * 16 * 7; idx += 256) {
        const int br  = idx / 112;
        const int rem = idx % 112;
        const int o   = rem / 7;
        const int p   = rem % 7;
        const float* wc = (br == 0) ? w1c : ((br == 1) ? w2c : w3c);
        const float* bc = (br == 0) ? b1c : ((br == 1) ? b2c : b3c);
        float acc = bc[o];
        #pragma unroll
        for (int ci = 0; ci < 2; ++ci)
            #pragma unroll
            for (int t = 0; t < 4; ++t) {
                const int q = p - 1 + t;
                if (q >= 0 && q < 8)
                    acc += wc[o * 8 + ci * 4 + t] * in0[ci * 32 + 8 * br + q];
            }
        s_conv[br][o][p] = fmaxf(acc, 0.0f);
    }
    __syncthreads();

    // ---- phase 0b: fused maxpool2 + Linear(48,16) -> s_bout ----
    if (tid < 48) {
        const int br = tid / 16;
        const int j  = tid % 16;
        const float* wo = (br == 0) ? w1o : ((br == 1) ? w2o : w3o);
        const float* bo = (br == 0) ? b1o : ((br == 1) ? b2o : b3o);
        float acc = bo[j];
        #pragma unroll
        for (int o2 = 0; o2 < 16; ++o2)
            #pragma unroll
            for (int m = 0; m < 3; ++m) {
                const float pooled = fmaxf(s_conv[br][o2][2 * m], s_conv[br][o2][2 * m + 1]);
                acc += wo[j * 48 + o2 * 3 + m] * pooled;
            }
        s_bout[br][j] = acc;
    }
    __syncthreads();

    // ---- phase 0c: assemble x[2,32] = [c1|c2|c3|inputs[:,24:]] ----
    if (tid < 64) {
        const int r  = tid >> 5;
        const int jc = tid & 31;
        float v;
        if (jc < 24) { const int br = jc >> 3, cc = jc & 7; v = s_bout[br][r * 8 + cc]; }
        else         { v = in0[r * 32 + jc]; }
        s_x[r][jc] = v;
    }
    __syncthreads();

    // ---- phase 0d: BatchNorm1d (train mode, batch=2) -> f16 A tile rows 0,1 ----
    if (tid < 32) {
        const int j = tid;
        const float x0 = s_x[0][j], x1 = s_x[1][j];
        const float mn  = 0.5f * (x0 + x1);
        const float dev = 0.5f * (x0 - x1);
        const float inv = rsqrtf(dev * dev + BN_EPS);
        s_actA[0][j] = (_Float16)(g0[j] * (x0 - mn) * inv + be0[j]);
        s_actA[1][j] = (_Float16)(g0[j] * (x1 - mn) * inv + be0[j]);
    }
    __syncthreads();

    // ---- layer 1: h1 = tanh(x @ W1^T + bl1)  (one WMMA per 16-col tile) ----
    {
        const int n = wv * 16 + mrow;
        v16h a, bf;
        v8f  c = {};
        #pragma unroll
        for (int j = 0; j < 16; ++j) a[j]  = s_actA[mrow][kmap(grp, j)];
        #pragma unroll
        for (int j = 0; j < 16; ++j) bf[j] = (_Float16)W1[n * 32 + kmap(grp, j)];
        c = __builtin_amdgcn_wmma_f32_16x16x32_f16(false, a, false, bf, (short)0, c, false, false);
        // only rows M=0,1 carry real data (padding rows are permanently zero in LDS)
        if (grp == 0) {
            const float bias = bl1[n];
            s_actB[0][n] = (_Float16)tanhf(c[0] + bias);
            s_actB[1][n] = (_Float16)tanhf(c[1] + bias);
        }
    }
    __syncthreads();

    // ---- layer 2: h2 = tanh(h1 @ W2^T + bl2)  (4 chained WMMAs per tile) ----
    {
        const int n = wv * 16 + mrow;
        v8f c = {};
        #pragma unroll
        for (int kc = 0; kc < 4; ++kc) {
            const int kb = kc * 32;
            v16h a, bf;
            #pragma unroll
            for (int j = 0; j < 16; ++j) a[j]  = s_actB[mrow][kb + kmap(grp, j)];
            #pragma unroll
            for (int j = 0; j < 16; ++j) bf[j] = (_Float16)W2[n * 128 + kb + kmap(grp, j)];
            c = __builtin_amdgcn_wmma_f32_16x16x32_f16(false, a, false, bf, (short)0, c, false, false);
        }
        if (grp == 0) {
            const float bias = bl2[n];
            s_actA[0][n] = (_Float16)tanhf(c[0] + bias);
            s_actA[1][n] = (_Float16)tanhf(c[1] + bias);
        }
    }
    __syncthreads();

    // ---- heads: [V | mu | L] = h2 @ [Vw;Mw;Lw]^T + bias  (73 cols, 5 tiles) ----
    if (wv < 5) {
        const int n = wv * 16 + mrow;               // 0..79 (73..79 padded)
        const float* wrow = nullptr;
        float bias = 0.0f;
        if (n == 0)       { wrow = Vw;                  bias = Vb[0];     }
        else if (n < 9)   { wrow = Mw + (n - 1) * 128;  bias = Mb[n - 1]; }
        else if (n < 73)  { wrow = Lw + (n - 9) * 128;  bias = Lb[n - 9]; }
        v8f c = {};
        #pragma unroll
        for (int kc = 0; kc < 4; ++kc) {
            const int kb = kc * 32;
            v16h a, bf;
            #pragma unroll
            for (int j = 0; j < 16; ++j) a[j] = s_actA[mrow][kb + kmap(grp, j)];
            #pragma unroll
            for (int j = 0; j < 16; ++j)
                bf[j] = wrow ? (_Float16)wrow[kb + kmap(grp, j)] : (_Float16)0.0f;
            c = __builtin_amdgcn_wmma_f32_16x16x32_f16(false, a, false, bf, (short)0, c, false, false);
        }
        if (grp == 0) {                              // rows M=0,1 live in c[0],c[1] of lanes 0..15
            s_h3[0][n] = c[0] + bias;
            s_h3[1][n] = c[1] + bias;
        }
    }
    __syncthreads();

    // ---- NAF epilogue: mu, V, Q = V - 0.5*||L^T (u - mu)||^2 ----
    if (tid < 2) {
        const int bb = tid;
        const float V = s_h3[bb][0];
        float mu[8], d[8];
        #pragma unroll
        for (int i = 0; i < 8; ++i) {
            mu[i] = tanhf(s_h3[bb][1 + i]);
            d[i]  = u[bb * 8 + i] - mu[i];
        }
        float acc = 0.0f;
        #pragma unroll
        for (int j = 0; j < 8; ++j) {
            float s = 0.0f;
            #pragma unroll
            for (int i = 0; i < 8; ++i) {
                const float raw = s_h3[bb][9 + i * 8 + j];
                const float Lij = (i > j) ? raw : ((i == j) ? expf(raw) : 0.0f);
                s += Lij * d[i];
            }
            acc += s * s;
        }
        const float Q = V - 0.5f * acc;
        #pragma unroll
        for (int j = 0; j < 8; ++j) out[bb * 8 + j] = mu[j];
        out[16 + bb] = Q;
        out[18 + bb] = V;
    }
}

extern "C" void kernel_launch(void* const* d_in, const int* in_sizes, int n_in,
                              void* d_out, int out_size, void* d_ws, size_t ws_size,
                              hipStream_t stream) {
    (void)in_sizes; (void)n_in; (void)out_size; (void)d_ws; (void)ws_size;
    naf_policy_kernel<<<1, 256, 0, stream>>>(
        (const float*)d_in[0],  (const float*)d_in[1],
        (const float*)d_in[2],  (const float*)d_in[3],
        (const float*)d_in[4],  (const float*)d_in[5],
        (const float*)d_in[6],  (const float*)d_in[7],
        (const float*)d_in[8],  (const float*)d_in[9],
        (const float*)d_in[10], (const float*)d_in[11],
        (const float*)d_in[12], (const float*)d_in[13],
        (const float*)d_in[14], (const float*)d_in[15],
        (const float*)d_in[16], (const float*)d_in[17],
        (const float*)d_in[18], (const float*)d_in[19],
        (const float*)d_in[20], (const float*)d_in[21],
        (const float*)d_in[22], (const float*)d_in[23],
        (const float*)d_in[24], (const float*)d_in[25],
        (float*)d_out);
}
